// GeometricLinear_35296041238922
// MI455X (gfx1250) — compile-verified, matrix operand order checked by reading, not run on hardware
//
#include <hip/hip_runtime.h>
#include <hip/hip_bf16.h>
#include <stdint.h>

// ---------------------------------------------------------------------------
// GeometricLinear (Cl(3,0)) as one 4096x4096x4096 GEMM on CDNA5 WMMA (bf16
// split-precision, f32 accumulate), double-buffered LDS with async
// global->LDS copies (GLOBAL_LOAD_ASYNC_TO_LDS_B128, ASYNCcnt path).
// ---------------------------------------------------------------------------

typedef __attribute__((ext_vector_type(16))) __bf16 v16bf;
typedef __attribute__((ext_vector_type(8)))  float  v8f;
typedef __attribute__((ext_vector_type(4)))  int    v4i;

#define BATCH 4096
#define KDIM  4096   // IN_F * 8
#define NDIM  4096   // OUT_F * 8
#define KT    32     // K tile (one bf16 WMMA K)
#define NKT   (KDIM / KT)

// --- async global->LDS b128 copy -------------------------------------------
// Probe-confirmed prototype (from clang diagnostic): arg0 = int4 AS1*,
// arg1 = int4 AS3*, then imm offset, imm cpol.
#define AS1V4(p) ((__attribute__((address_space(1))) v4i*)(uintptr_t)(p))
#define AS3V4(p) ((__attribute__((address_space(3))) v4i*)(uint32_t)(uintptr_t)(p))

#if __has_builtin(__builtin_amdgcn_global_load_async_to_lds_b128)
  #define ASYNC_COPY_B128(g, l) \
      __builtin_amdgcn_global_load_async_to_lds_b128(AS1V4(g), AS3V4(l), 0, 0)
  #define HAVE_ASYNC 1
#else
  #define ASYNC_COPY_B128(g, l) (*(uint4*)(l) = *(const uint4*)(g))
  #define HAVE_ASYNC 0
#endif

#if HAVE_ASYNC
  #if __has_builtin(__builtin_amdgcn_s_wait_asynccnt)
    #define WAIT_ASYNC() __builtin_amdgcn_s_wait_asynccnt(0)
  #else
    #define WAIT_ASYNC() asm volatile("s_wait_asynccnt 0" ::: "memory")
  #endif
#else
  #define WAIT_ASYNC() ((void)0)
#endif

// Cayley signed-permutation tables: (a*b)[r] = sum_p a[p] * SGN[p][r] * b[QIDX[p][r]]
__constant__ int QIDX[8][8] = {
  {0,1,2,3,4,5,6,7},
  {1,0,4,5,2,3,7,6},
  {2,4,0,6,1,7,3,5},
  {3,5,6,0,7,1,2,4},
  {4,2,1,7,0,6,5,3},
  {5,3,7,1,6,0,4,2},
  {6,7,3,2,5,4,0,1},
  {7,6,5,4,3,2,1,0}};
__constant__ float SGN[8][8] = {
  {+1,+1,+1,+1,+1,+1,+1,+1},
  {+1,+1,+1,+1,+1,+1,+1,+1},
  {+1,-1,+1,+1,-1,-1,+1,-1},
  {+1,-1,-1,+1,+1,-1,-1,+1},
  {-1,+1,-1,-1,+1,+1,-1,+1},
  {-1,+1,+1,-1,-1,+1,+1,-1},
  {-1,-1,+1,-1,+1,-1,+1,+1},
  {-1,-1,+1,-1,+1,-1,+1,+1}};

// ---- pre-pass: x (f32) -> bf16 hi/lo pair ---------------------------------
__global__ __launch_bounds__(256) void geo_prep_x(const float* __restrict__ x,
                                                  __bf16* __restrict__ xhi,
                                                  __bf16* __restrict__ xlo) {
  size_t t = (size_t)blockIdx.x * blockDim.x + threadIdx.x;
  float v = x[t];
  __bf16 h = (__bf16)v;
  xhi[t] = h;
  xlo[t] = (__bf16)(v - (float)h);
}

// ---- pre-pass: weight -> Wt[n][k] (bf16 hi/lo), n=i*8+r, k=j*8+p ----------
__global__ __launch_bounds__(256) void geo_prep_w(const float* __restrict__ w,
                                                  __bf16* __restrict__ whi,
                                                  __bf16* __restrict__ wlo) {
  int t = blockIdx.x * blockDim.x + threadIdx.x;   // 0 .. 512*512-1
  int i = t >> 9, j = t & 511;
  float wq[8];
  const float* wp = w + (size_t)t * 8;
#pragma unroll
  for (int q = 0; q < 8; ++q) wq[q] = wp[q];

#pragma unroll
  for (int r = 0; r < 8; ++r) {
    union { __bf16 h[8]; uint4 q; } uh, ul;
#pragma unroll
    for (int p = 0; p < 8; ++p) {
      float v = SGN[p][r] * wq[QIDX[p][r]];
      __bf16 h = (__bf16)v;
      uh.h[p] = h;
      ul.h[p] = (__bf16)(v - (float)h);
    }
    size_t n = (size_t)i * 8 + r;
    size_t k = (size_t)j * 8;
    *(uint4*)(whi + n * KDIM + k) = uh.q;
    *(uint4*)(wlo + n * KDIM + k) = ul.q;
  }
}

// ---- WMMA GEMM ------------------------------------------------------------
__device__ inline v8f wmma_bf16(v16bf a, v16bf b, v8f c) {
  return __builtin_amdgcn_wmma_f32_16x16x32_bf16(false, a, false, b,
                                                 (short)0, c, false, false);
}

// lane L (half = L/16) holds K[8h..8h+7] in v0..v3 and K[16+8h..] in v4..v7.
__device__ inline v16bf load_frag(const __bf16* rowbase, int half) {
  union { v16bf v; uint4 q[2]; } u;
  u.q[0] = *(const uint4*)(rowbase + 8 * half);
  u.q[1] = *(const uint4*)(rowbase + 16 + 8 * half);
  return u.v;
}

// Stage one 128x32 tile set (Ah, Al, Bh, Bl) into LDS stage buffer.
__device__ inline void stage_tiles(const __bf16* __restrict__ Xhi,
                                   const __bf16* __restrict__ Xlo,
                                   const __bf16* __restrict__ Whi,
                                   const __bf16* __restrict__ Wlo,
                                   __bf16* sAh, __bf16* sAl,
                                   __bf16* sBh, __bf16* sBl,
                                   int tid, int m0, int n0, int k0) {
#pragma unroll
  for (int s = 0; s < 2; ++s) {
    int c   = tid + s * 256;        // 512 chunks of 8 bf16 per array
    int row = c >> 2;
    int col = (c & 3) << 3;
    size_t ga = (size_t)(m0 + row) * KDIM + (k0 + col);
    size_t gb = (size_t)(n0 + row) * KDIM + (k0 + col);
    int lo = row * 32 + col;
    ASYNC_COPY_B128(&Xhi[ga], &sAh[lo]);
    ASYNC_COPY_B128(&Xlo[ga], &sAl[lo]);
    ASYNC_COPY_B128(&Whi[gb], &sBh[lo]);
    ASYNC_COPY_B128(&Wlo[gb], &sBl[lo]);
  }
}

__global__ __launch_bounds__(256) void geo_gemm(const __bf16* __restrict__ Xhi,
                                                const __bf16* __restrict__ Xlo,
                                                const __bf16* __restrict__ Whi,
                                                const __bf16* __restrict__ Wlo,
                                                const float* __restrict__ bias,
                                                float* __restrict__ out) {
  // [stage][Ah,Al,Bh,Bl][128*32]  -> 2 * 32KB = 64KB
  __shared__ __bf16 smem[2][4][128 * 32];

  const int tid  = threadIdx.x;
  const int wave = tid >> 5;
  const int lane = tid & 31;
  const int wm   = wave >> 2;       // 0..1  -> 64-row slab
  const int wn   = wave & 3;        // 0..3  -> 32-col slab
  const int half = lane >> 4;
  const int l16  = lane & 15;

  const int m0 = blockIdx.y * 128;
  const int n0 = blockIdx.x * 128;

  v8f acc[4][2] = {};

  // preload stage 0
  stage_tiles(Xhi, Xlo, Whi, Wlo,
              smem[0][0], smem[0][1], smem[0][2], smem[0][3],
              tid, m0, n0, 0);

  for (int kt = 0; kt < NKT; ++kt) {
    const int cur = kt & 1;
    WAIT_ASYNC();        // my async copies into `cur` have landed in LDS
    __syncthreads();     // everyone's copies landed; everyone done reading nxt

    if (kt + 1 < NKT)    // prefetch next K tile into the other stage
      stage_tiles(Xhi, Xlo, Whi, Wlo,
                  smem[cur ^ 1][0], smem[cur ^ 1][1],
                  smem[cur ^ 1][2], smem[cur ^ 1][3],
                  tid, m0, n0, (kt + 1) * KT);

    const __bf16* cAh = smem[cur][0];
    const __bf16* cAl = smem[cur][1];
    const __bf16* cBh = smem[cur][2];
    const __bf16* cBl = smem[cur][3];

    v16bf ah[4], al[4], bh[2], bl[2];
#pragma unroll
    for (int mt = 0; mt < 4; ++mt) {
      int row = (wm * 64 + mt * 16 + l16) * 32;
      ah[mt] = load_frag(cAh + row, half);
      al[mt] = load_frag(cAl + row, half);
    }
#pragma unroll
    for (int nt = 0; nt < 2; ++nt) {
      int row = (wn * 32 + nt * 16 + l16) * 32;
      bh[nt] = load_frag(cBh + row, half);
      bl[nt] = load_frag(cBl + row, half);
    }

#pragma unroll
    for (int mt = 0; mt < 4; ++mt)
#pragma unroll
      for (int nt = 0; nt < 2; ++nt) {
        v8f c = acc[mt][nt];
        c = wmma_bf16(ah[mt], bh[nt], c);  // hi*hi
        c = wmma_bf16(ah[mt], bl[nt], c);  // hi*lo
        c = wmma_bf16(al[mt], bh[nt], c);  // lo*hi  (lo*lo negligible)
        acc[mt][nt] = c;
      }
  }

  // Epilogue: lane -> n=l16, m = vgpr + 8*half. bias indexed by GEMM n.
#pragma unroll
  for (int mt = 0; mt < 4; ++mt)
#pragma unroll
    for (int nt = 0; nt < 2; ++nt) {
      int n  = n0 + wn * 32 + nt * 16 + l16;
      float bv = bias[n];
      int mb = m0 + wm * 64 + mt * 16 + 8 * half;
      v8f a = acc[mt][nt];
#pragma unroll
      for (int v = 0; v < 8; ++v)
        out[(size_t)(mb + v) * NDIM + n] = a[v] + bv;
    }
}

// ---------------------------------------------------------------------------
extern "C" void kernel_launch(void* const* d_in, const int* in_sizes, int n_in,
                              void* d_out, int out_size, void* d_ws, size_t ws_size,
                              hipStream_t stream) {
  (void)in_sizes; (void)n_in; (void)out_size; (void)ws_size;
  const float* x    = (const float*)d_in[0];   // [4096, 512, 8]
  const float* w    = (const float*)d_in[1];   // [512, 512, 8]
  const float* bias = (const float*)d_in[2];   // [512, 8] == flat [4096] over n
  float* out = (float*)d_out;                  // [4096][4096]

  char* ws = (char*)d_ws;
  const size_t SZ = (size_t)BATCH * KDIM * sizeof(__bf16);  // 32 MiB per array
  __bf16* Xhi = (__bf16*)(ws + 0 * SZ);
  __bf16* Xlo = (__bf16*)(ws + 1 * SZ);
  __bf16* Whi = (__bf16*)(ws + 2 * SZ);
  __bf16* Wlo = (__bf16*)(ws + 3 * SZ);

  geo_prep_x<<<(BATCH * KDIM) / 256, 256, 0, stream>>>(x, Xhi, Xlo);
  geo_prep_w<<<(512 * 512) / 256, 256, 0, stream>>>(w, Whi, Wlo);

  dim3 grid(NDIM / 128, BATCH / 128);
  geo_gemm<<<grid, 256, 0, stream>>>(Xhi, Xlo, Whi, Wlo, bias, out);
}
